// SimpleGateLoopLayer_71408126263386
// MI455X (gfx1250) — matrix operand order repaired and checked by reading, not compiled, not run on hardware
//
#include <hip/hip_runtime.h>
#include <hip/hip_bf16.h>

typedef __attribute__((ext_vector_type(16))) __bf16 v16bf;
typedef __attribute__((ext_vector_type(8)))  __bf16 v8bf;
typedef __attribute__((ext_vector_type(8)))  float  v8f;
typedef int v4i __attribute__((vector_size(16)));   // matches async-LDS builtin param type

#define DIM   1024
#define SEQ   4096
#define BATCH 4
#define MROWS (BATCH * SEQ)   // 16384
#define NCOLS (3 * DIM)       // 3072
#define BK    32
#define NK    (DIM / BK)      // 32 K-steps
#define LDT   40              // padded LDS row stride (elements): 80B, conflict-free & 16B-aligned
#define NCHUNK 16
#define CHUNK  256            // SEQ / NCHUNK

// ---- CDNA5 async global->LDS path (guarded; falls back to sync copies) ----
#if defined(__has_builtin)
#if __has_builtin(__builtin_amdgcn_global_load_async_to_lds_b128) && \
    __has_builtin(__builtin_amdgcn_s_wait_asynccnt)
#define USE_ASYNC_LDS 1
#endif
#endif
#ifndef USE_ASYNC_LDS
#define USE_ASYNC_LDS 0
#endif

#define AS1 __attribute__((address_space(1)))
#define AS3 __attribute__((address_space(3)))

__device__ __forceinline__ void copy16(const __bf16* g, __bf16* l) {
#if USE_ASYNC_LDS
    __builtin_amdgcn_global_load_async_to_lds_b128((AS1 v4i*)g, (AS3 v4i*)l, 0, 0);
#else
    *(uint4*)l = *(const uint4*)g;
#endif
}

// ---------------------------------------------------------------------------
// Kernel 1: RMSNorm + bf16 hi/lo split of xn = x/||x|| * sqrt(d) * gamma
// ---------------------------------------------------------------------------
__global__ __launch_bounds__(256)
void rmsnorm_split(const float* __restrict__ x, const float* __restrict__ gamma,
                   __bf16* __restrict__ xh, __bf16* __restrict__ xl) {
    __shared__ float red[256];
    const int row = blockIdx.x;
    const int t   = threadIdx.x;
    const float4* xr = (const float4*)(x + (size_t)row * DIM);
    float4 v = xr[t];
    red[t] = v.x * v.x + v.y * v.y + v.z * v.z + v.w * v.w;
    __syncthreads();
    for (int s = 128; s > 0; s >>= 1) {
        if (t < s) red[t] += red[t + s];
        __syncthreads();
    }
    const float scale = 32.0f / fmaxf(sqrtf(red[0]), 1e-12f);  // sqrt(1024)=32
    const float4 gv = ((const float4*)gamma)[t];
    float o[4] = { v.x * scale * gv.x, v.y * scale * gv.y,
                   v.z * scale * gv.z, v.w * scale * gv.w };
    __bf16 hs[4], ls[4];
#pragma unroll
    for (int i = 0; i < 4; ++i) {
        hs[i] = (__bf16)o[i];
        ls[i] = (__bf16)(o[i] - (float)hs[i]);
    }
    *(uint2*)(xh + (size_t)row * DIM + t * 4) = *(uint2*)hs;
    *(uint2*)(xl + (size_t)row * DIM + t * 4) = *(uint2*)ls;
}

// ---------------------------------------------------------------------------
// Kernel 2: bf16 hi/lo split of W (3072x1024)
// ---------------------------------------------------------------------------
__global__ __launch_bounds__(256)
void wsplit(const float* __restrict__ w, __bf16* __restrict__ wh, __bf16* __restrict__ wl) {
    const size_t i = (size_t)blockIdx.x * 256 + threadIdx.x;  // group of 4 elements
    float4 v = ((const float4*)w)[i];
    float o[4] = { v.x, v.y, v.z, v.w };
    __bf16 hs[4], ls[4];
#pragma unroll
    for (int k = 0; k < 4; ++k) {
        hs[k] = (__bf16)o[k];
        ls[k] = (__bf16)(o[k] - (float)hs[k]);
    }
    *(uint2*)(wh + i * 4) = *(uint2*)hs;
    *(uint2*)(wl + i * 4) = *(uint2*)ls;
}

// ---------------------------------------------------------------------------
// Kernel 3: WMMA GEMM  h[m,e] = sum_k xn[m,k]*W[e,k]  with bf16x3 split-precision.
// Block tile 128x128, 8 waves -> each wave 32(M) x 64(N) = 2x4 WMMA tiles.
// Double-buffered LDS (2 x 40KB), async global->LDS pipeline on ASYNCcnt.
// Epilogue: e<1024 -> q (d_out); 1024..2047 -> kv; 2048.. -> sigmoid -> a.
// ---------------------------------------------------------------------------
__global__ __launch_bounds__(256)
void gemm_qkva(const __bf16* __restrict__ xh, const __bf16* __restrict__ xl,
               const __bf16* __restrict__ wh, const __bf16* __restrict__ wl,
               float* __restrict__ qout, float* __restrict__ kvout, float* __restrict__ aout) {
    __shared__ __align__(16) __bf16 sAh[2][128 * LDT];
    __shared__ __align__(16) __bf16 sAl[2][128 * LDT];
    __shared__ __align__(16) __bf16 sBh[2][128 * LDT];
    __shared__ __align__(16) __bf16 sBl[2][128 * LDT];

    const int tid = threadIdx.x;
    const int m0  = blockIdx.x * 128;
    const int n0  = blockIdx.y * 128;

    // global->LDS staging role: 2 threads per row, 16 elements (32B) each
    const int lrow = tid >> 1;
    const int lseg = (tid & 1) * 16;
    const int ls   = lrow * LDT + lseg;
    const size_t gaOff = (size_t)(m0 + lrow) * DIM + lseg;
    const size_t gbOff = (size_t)(n0 + lrow) * DIM + lseg;

    const int lane = tid & 31;
    const int wave = tid >> 5;
    const int wm   = (wave >> 1) * 32;   // 0,32,64,96
    const int wn   = (wave & 1) * 64;    // 0,64
    const int hl   = lane >> 4;          // half-wave select
    const int l15  = lane & 15;

    // WMMA fragment gather offsets (ISA 16-bit A/B layouts):
    // A lane<16: K{0..7,16..23}; lane>=16: K{8..15,24..31}
    // B lane<16: K{0..7,16..23}; lane>=16: K{8..15,24..31}
    const int aoff = l15 * LDT + hl * 8;   // second chunk at +16
    const int boff = l15 * LDT + hl * 8;   // second chunk at +16

    v8f acc[2][4];
#pragma unroll
    for (int mt = 0; mt < 2; ++mt)
#pragma unroll
        for (int nt = 0; nt < 4; ++nt) acc[mt][nt] = (v8f)(0.0f);

    // stage a 32-wide K slab into LDS buffer b (8 x 16B per thread)
    auto stage = [&](int kk, int b) {
        const __bf16* ah = xh + gaOff + kk;
        const __bf16* al = xl + gaOff + kk;
        const __bf16* bh = wh + gbOff + kk;
        const __bf16* bl = wl + gbOff + kk;
        copy16(ah, &sAh[b][ls]);  copy16(ah + 8, &sAh[b][ls + 8]);
        copy16(al, &sAl[b][ls]);  copy16(al + 8, &sAl[b][ls + 8]);
        copy16(bh, &sBh[b][ls]);  copy16(bh + 8, &sBh[b][ls + 8]);
        copy16(bl, &sBl[b][ls]);  copy16(bl + 8, &sBl[b][ls + 8]);
    };

    stage(0, 0);

    for (int k = 0; k < NK; ++k) {
        const int cur = k & 1;
        if (k + 1 < NK) {
            stage((k + 1) * BK, cur ^ 1);   // prefetch next slab into other buffer
#if USE_ASYNC_LDS
            __builtin_amdgcn_s_wait_asynccnt(8);  // my stage-k copies done (in-order)
#endif
        } else {
#if USE_ASYNC_LDS
            __builtin_amdgcn_s_wait_asynccnt(0);
#endif
        }
        __syncthreads();  // all waves' stage-k copies visible

        v16bf afh[2], afl[2], bfh[4], bfl[4];
#pragma unroll
        for (int mt = 0; mt < 2; ++mt) {
            const int base = aoff + (wm + mt * 16) * LDT;
            ((v8bf*)&afh[mt])[0] = *(const v8bf*)(&sAh[cur][base]);
            ((v8bf*)&afh[mt])[1] = *(const v8bf*)(&sAh[cur][base + 16]);
            ((v8bf*)&afl[mt])[0] = *(const v8bf*)(&sAl[cur][base]);
            ((v8bf*)&afl[mt])[1] = *(const v8bf*)(&sAl[cur][base + 16]);
        }
#pragma unroll
        for (int nt = 0; nt < 4; ++nt) {
            const int base = boff + (wn + nt * 16) * LDT;
            ((v8bf*)&bfh[nt])[0] = *(const v8bf*)(&sBh[cur][base]);
            ((v8bf*)&bfh[nt])[1] = *(const v8bf*)(&sBh[cur][base + 16]);
            ((v8bf*)&bfl[nt])[0] = *(const v8bf*)(&sBl[cur][base]);
            ((v8bf*)&bfl[nt])[1] = *(const v8bf*)(&sBl[cur][base + 16]);
        }
#pragma unroll
        for (int mt = 0; mt < 2; ++mt)
#pragma unroll
            for (int nt = 0; nt < 4; ++nt) {
                acc[mt][nt] = __builtin_amdgcn_wmma_f32_16x16x32_bf16(
                    false, afh[mt], false, bfh[nt], (short)0, acc[mt][nt], false, false);
                acc[mt][nt] = __builtin_amdgcn_wmma_f32_16x16x32_bf16(
                    false, afh[mt], false, bfl[nt], (short)0, acc[mt][nt], false, false);
                acc[mt][nt] = __builtin_amdgcn_wmma_f32_16x16x32_bf16(
                    false, afl[mt], false, bfh[nt], (short)0, acc[mt][nt], false, false);
            }
        __syncthreads();  // all waves done reading buf[cur] before it is re-staged
    }

    // Epilogue: segment is block-uniform (1024 % 128 == 0)
    const int seg = n0 >> 10;
    float* dst = (seg == 0) ? qout : (seg == 1) ? kvout : aout;
    const int colbase = (n0 & 1023) + wn + l15;
#pragma unroll
    for (int mt = 0; mt < 2; ++mt)
#pragma unroll
        for (int nt = 0; nt < 4; ++nt)
#pragma unroll
            for (int r = 0; r < 8; ++r) {
                const int mm = m0 + wm + mt * 16 + r + hl * 8;
                float v = acc[mt][nt][r];
                if (seg == 2) v = __builtin_amdgcn_rcpf(1.0f + __expf(-v));  // sigmoid(a)
                dst[(size_t)mm * DIM + colbase + nt * 16] = v;
            }
}

// ---------------------------------------------------------------------------
// Kernel 4: per-chunk local scan: A = prod(a), Y = local recurrence end state
// ---------------------------------------------------------------------------
__global__ __launch_bounds__(256)
void scan_local(const float* __restrict__ aw, const float* __restrict__ kv,
                float* __restrict__ cA, float* __restrict__ cY) {
    const int ch    = blockIdx.x * 256 + threadIdx.x;  // channel 0..1023
    const int chunk = blockIdx.y;
    const int batch = blockIdx.z;
    const size_t base = ((size_t)batch * SEQ + chunk * CHUNK) * DIM + ch;
    float A = 1.0f, Y = 0.0f;
    for (int t = 0; t < CHUNK; ++t) {
        const float at = aw[base + (size_t)t * DIM];
        const float kt = kv[base + (size_t)t * DIM];
        A *= at;
        Y = at * Y + kt;
    }
    const int ci = ((batch * NCHUNK + chunk) << 10) + ch;
    cA[ci] = A;
    cY[ci] = Y;
}

// ---------------------------------------------------------------------------
// Kernel 5: scan the 16 chunk carries per (batch, channel) -> carry-in per chunk
// ---------------------------------------------------------------------------
__global__ __launch_bounds__(256)
void scan_carry(const float* __restrict__ cA, const float* __restrict__ cY,
                float* __restrict__ cI) {
    const int g = blockIdx.x * 256 + threadIdx.x;  // 0..4095
    const int batch = g >> 10;
    const int ch = g & 1023;
    float y = 0.0f;
    for (int c = 0; c < NCHUNK; ++c) {
        const int idx = ((batch * NCHUNK + c) << 10) + ch;
        cI[idx] = y;
        y = cA[idx] * y + cY[idx];
    }
}

// ---------------------------------------------------------------------------
// Kernel 6: replay each chunk with carry-in, out = q * y  (q already in d_out)
// ---------------------------------------------------------------------------
__global__ __launch_bounds__(256)
void scan_apply(const float* __restrict__ aw, const float* __restrict__ kv,
                const float* __restrict__ cI, float* __restrict__ out) {
    const int ch    = blockIdx.x * 256 + threadIdx.x;
    const int chunk = blockIdx.y;
    const int batch = blockIdx.z;
    float y = cI[((batch * NCHUNK + chunk) << 10) + ch];
    const size_t base = ((size_t)batch * SEQ + chunk * CHUNK) * DIM + ch;
    for (int t = 0; t < CHUNK; ++t) {
        const size_t idx = base + (size_t)t * DIM;
        const float at = aw[idx];
        y = at * y + kv[idx];
        out[idx] = out[idx] * y;  // q * y
    }
}

// ---------------------------------------------------------------------------
extern "C" void kernel_launch(void* const* d_in, const int* in_sizes, int n_in,
                              void* d_out, int out_size, void* d_ws, size_t ws_size,
                              hipStream_t stream) {
    const float* x     = (const float*)d_in[0];
    const float* W     = (const float*)d_in[1];
    const float* gamma = (const float*)d_in[2];

    char* ws = (char*)d_ws;
    const size_t SZ_X16 = (size_t)MROWS * DIM * 2;   // 32 MB (bf16)
    const size_t SZ_W16 = (size_t)NCOLS * DIM * 2;   // 6 MB  (bf16)
    const size_t SZ_H32 = (size_t)MROWS * DIM * 4;   // 64 MB (f32)
    const size_t SZ_CR  = (size_t)BATCH * NCHUNK * DIM * 4;  // 256 KB

    __bf16* xh = (__bf16*)(ws);
    __bf16* xl = (__bf16*)(ws + SZ_X16);
    __bf16* wh = (__bf16*)(ws + 2 * SZ_X16);
    __bf16* wl = (__bf16*)(ws + 2 * SZ_X16 + SZ_W16);
    float*  kvb = (float*)(ws + 2 * SZ_X16 + 2 * SZ_W16);
    float*  ab  = (float*)(ws + 2 * SZ_X16 + 2 * SZ_W16 + SZ_H32);
    float*  cA  = (float*)(ws + 2 * SZ_X16 + 2 * SZ_W16 + 2 * SZ_H32);
    float*  cY  = (float*)(ws + 2 * SZ_X16 + 2 * SZ_W16 + 2 * SZ_H32 + SZ_CR);
    float*  cI  = (float*)(ws + 2 * SZ_X16 + 2 * SZ_W16 + 2 * SZ_H32 + 2 * SZ_CR);

    float* q = (float*)d_out;

    rmsnorm_split<<<MROWS, 256, 0, stream>>>(x, gamma, xh, xl);
    wsplit<<<(NCOLS * DIM) / (256 * 4), 256, 0, stream>>>(W, wh, wl);
    gemm_qkva<<<dim3(MROWS / 128, NCOLS / 128), 256, 0, stream>>>(xh, xl, wh, wl, q, kvb, ab);
    scan_local<<<dim3(DIM / 256, NCHUNK, BATCH), 256, 0, stream>>>(ab, kvb, cA, cY);
    scan_carry<<<(BATCH * DIM) / 256, 256, 0, stream>>>(cA, cY, cI);
    scan_apply<<<dim3(DIM / 256, NCHUNK, BATCH), 256, 0, stream>>>(ab, kvb, cI, q);
}
